// PhysicsAttention_38680475468398
// MI455X (gfx1250) — matrix operand order repaired and checked by reading, not compiled, not run on hardware
//
#include <hip/hip_runtime.h>

typedef __attribute__((ext_vector_type(16))) __bf16 v16bf;
typedef __attribute__((ext_vector_type(8)))  float  v8f;
typedef __attribute__((ext_vector_type(4)))  unsigned int v4u;
typedef __attribute__((ext_vector_type(8)))  int v8i;
typedef __attribute__((ext_vector_type(4)))  int v4i;

#define DEVINL __device__ __forceinline__

constexpr int Bb = 4, Nn = 16384, Cc = 256, Ss = 64, Hh = 8;

#if defined(__has_builtin)
#if __has_builtin(__builtin_amdgcn_tensor_load_to_lds)
#define HAVE_TDM 1
#endif
#endif
#ifndef HAVE_TDM
#define HAVE_TDM 0
#endif

DEVINL unsigned short f2bf(float x) {
  union { __bf16 b; unsigned short u; } c;
  c.b = (__bf16)x;
  return c.u;
}

DEVINL v8f vzero8() {
  v8f z;
#pragma unroll
  for (int i = 0; i < 8; ++i) z[i] = 0.f;
  return z;
}

DEVINL v16bf ones_frag() {
  union { v16bf v; unsigned short u[16]; } o;
#pragma unroll
  for (int i = 0; i < 16; ++i) o.u[i] = 0x3F80;   // bf16 1.0
  return o.v;
}

// A-frag (16x32 bf16, row-major source, leading dim ldk):
// lane holds row = lane&15; elements i: K = 16*(i>>3) + 8*(lane>>4) + (i&7)
DEVINL v16bf load_afrag(const unsigned short* base, int ldk) {
  const int lane = threadIdx.x & 31;
  const unsigned short* r = base + (lane & 15) * ldk + ((lane >> 4) << 3);
  union { v16bf v; uint4 q[2]; } f;
  f.q[0] = *(const uint4*)(r);
  f.q[1] = *(const uint4*)(r + 16);
  return f.v;
}

// A-frag sourced from fp32 row-major, converting via native bf16 cvt.
DEVINL v16bf load_afrag_f32(const float* base, int ldk) {
  const int lane = threadIdx.x & 31;
  const float* r = base + (lane & 15) * ldk + ((lane >> 4) << 3);
  alignas(16) float t[16];
  *(float4*)&t[0]  = *(const float4*)(r);
  *(float4*)&t[4]  = *(const float4*)(r + 4);
  *(float4*)&t[8]  = *(const float4*)(r + 16);
  *(float4*)&t[12] = *(const float4*)(r + 20);
  v16bf v;
#pragma unroll
  for (int i = 0; i < 16; ++i) v[i] = (__bf16)t[i];
  return v;
}

// B-frag (32x16 bf16) from packed tile [col16][k32]: two 16B chunks per lane, K = 16*(i>>3) + 8*(lane>>4) + (i&7).
DEVINL v16bf load_bfrag(const unsigned short* tile) {
  const int lane = threadIdx.x & 31;
  const unsigned short* r = tile + (lane & 15) * 32 + ((lane >> 4) << 3);
  union { v16bf v; uint4 q[2]; } f;
  f.q[0] = *(const uint4*)(r);
  f.q[1] = *(const uint4*)(r + 16);
  return f.v;
}

DEVINL v8f wmma_bf16(v16bf a, v16bf b, v8f c) {
  return __builtin_amdgcn_wmma_f32_16x16x32_bf16(false, a, false, b, (short)0, c,
                                                 false, false);
}

#if HAVE_TDM
// TDM: async copy of n8*8 contiguous bytes global -> LDS (1-D tile, 8B elems).
// D# per ISA 8.3/8.4: group0 = {count=1, lds_addr, gaddr lo, gaddr hi | type=2}
DEVINL void tdm_load(unsigned lds_off, const void* gsrc, unsigned n8) {
  const unsigned long long g = (unsigned long long)gsrc;
  v4u g0 = {1u, lds_off, (unsigned)g,
            (unsigned)((g >> 32) & 0x01FFFFFFu) | 0x80000000u};
  v8i g1;
  g1[0] = (int)(3u << 16);                      // data_size = 8B
  g1[1] = (int)((n8 & 0xFFFFu) << 16);          // tensor_dim0 lo16
  g1[2] = (int)((n8 >> 16) | (1u << 16));       // tensor_dim0 hi16 | tensor_dim1=1
  g1[3] = (int)((n8 & 0xFFFFu) << 16);          // tile_dim0
  g1[4] = 0;
  g1[5] = (int)n8;                              // tensor_dim0_stride
  g1[6] = 0;
  g1[7] = 0;
  v4i z4 = {0, 0, 0, 0};
#if __clang_major__ >= 23
  v8i z8 = {0, 0, 0, 0, 0, 0, 0, 0};
  __builtin_amdgcn_tensor_load_to_lds(g0, g1, z4, z4, z8, 0);
#else
  __builtin_amdgcn_tensor_load_to_lds(g0, g1, z4, z4, 0);
#endif
}
#endif

// ---------------- k0: pack fp32 weight [Ktot x Ntot] -> bf16 B-frag tiles ----
__global__ void k0_pack(const float* __restrict__ W, unsigned short* __restrict__ out,
                        int Ktot, int Ntot) {
  const int kts = Ktot >> 5, nts = Ntot >> 4;
  const int total = nts * kts * 64;
  for (int task = blockIdx.x * blockDim.x + threadIdx.x; task < total;
       task += gridDim.x * blockDim.x) {
    const int k8 = task & 3, col = (task >> 2) & 15;
    const int kt = (task >> 6) % kts, nt = (task >> 6) / kts;
    alignas(16) unsigned short tmp[8];
#pragma unroll
    for (int j = 0; j < 8; ++j)
      tmp[j] = f2bf(W[(size_t)(kt * 32 + k8 * 8 + j) * Ntot + nt * 16 + col]);
    *(uint4*)(out + (((size_t)(nt * kts + kt) * 16 + col) * 32 + k8 * 8)) =
        *(const uint4*)tmp;
  }
}

// ---------------- k1: x->h, slice softmax, qkv (64 rows / block) -------------
__global__ __launch_bounds__(128) void k1_fused(
    const float* __restrict__ x, const float* __restrict__ b0,
    const float* __restrict__ bs, const float* __restrict__ bqkv,
    const unsigned short* __restrict__ W0p, const unsigned short* __restrict__ Wsp,
    const unsigned short* __restrict__ Wqkvp,
    unsigned short* __restrict__ swN, unsigned short* __restrict__ swAp,
    unsigned short* __restrict__ qkvP) {
  __shared__ unsigned short hA[64][256];             // 32 KB
  __shared__ float colT[64][16];                     //  4 KB
  __shared__ __align__(16) unsigned char smix[16384];// sws (16KB) | Wqkv panels x2
  float (*sws)[64] = (float(*)[64])smix;
  unsigned short (*wpanel)[4096] = (unsigned short(*)[4096])smix;
  const int wave = threadIdx.x >> 5, lane = threadIdx.x & 31;
  const int rows0 = blockIdx.x * 64;
  const int b = rows0 / Nn, n0 = rows0 % Nn;
  const int kt0 = n0 >> 5;
  const int wrow = wave * 16;
  const int mrow = (lane >> 4) << 3;

  // ---- h = x @ W0 + b0 ----
  {
    v8f acc[16];
#pragma unroll
    for (int nt = 0; nt < 16; ++nt) acc[nt] = vzero8();
    const float* xrow = x + (size_t)(rows0 + wrow) * Cc;
    for (int ks = 0; ks < 8; ++ks) {
      v16bf a = load_afrag_f32(xrow + ks * 32, Cc);
      v16bf bc = load_bfrag(W0p + ((size_t)ks << 9));
#pragma unroll
      for (int nt = 0; nt < 16; ++nt) {
        v16bf bn = bc;
        if (nt < 15) bn = load_bfrag(W0p + (((size_t)(nt + 1) * 8 + ks) << 9));
        acc[nt] = wmma_bf16(a, bc, acc[nt]);
        bc = bn;
      }
    }
#pragma unroll
    for (int nt = 0; nt < 16; ++nt) {
      const int col = nt * 16 + (lane & 15);
      const float bias = b0[col];
#pragma unroll
      for (int r = 0; r < 8; ++r)
        hA[wrow + r + mrow][col] = f2bf(acc[nt][r] + bias);
    }
  }
  __syncthreads();

  // ---- slice logits + softmax over S=64 ----
  {
    v8f sacc[4];
#pragma unroll
    for (int st = 0; st < 4; ++st) sacc[st] = vzero8();
    for (int ks = 0; ks < 8; ++ks) {
      v16bf a = load_afrag(&hA[wrow][ks * 32], 256);
      v16bf bc = load_bfrag(Wsp + ((size_t)ks << 9));
#pragma unroll
      for (int st = 0; st < 4; ++st) {
        v16bf bn = bc;
        if (st < 3) bn = load_bfrag(Wsp + (((size_t)(st + 1) * 8 + ks) << 9));
        sacc[st] = wmma_bf16(a, bc, sacc[st]);
        bc = bn;
      }
    }
#pragma unroll
    for (int st = 0; st < 4; ++st) {
      const int col = st * 16 + (lane & 15);
      const float bias = bs[col];
#pragma unroll
      for (int r = 0; r < 8; ++r)
        sws[wrow + r + mrow][col] = sacc[st][r] + bias;
    }
  }
  __syncthreads();
  if (threadIdx.x < 64) {
    float* row = sws[threadIdx.x];
    float mx = -1e30f;
    for (int j = 0; j < 64; ++j) mx = fmaxf(mx, row[j]);
    float sum = 0.f;
    for (int j = 0; j < 64; ++j) { float e = __expf(row[j] - mx); row[j] = e; sum += e; }
    const float inv = 1.f / sum;
    unsigned short* dst = swN + (size_t)(rows0 + threadIdx.x) * Ss;
    for (int j = 0; j < 64; ++j) { float v = row[j] * inv; row[j] = v; dst[j] = f2bf(v); }
  }
  __syncthreads();
  for (int task = threadIdx.x; task < 512; task += 128) {
    const int k8 = task & 3, s = (task >> 2) & 15, stile = (task >> 6) & 3, ktl = task >> 8;
    alignas(16) unsigned short tmp[8];
#pragma unroll
    for (int j = 0; j < 8; ++j)
      tmp[j] = f2bf(sws[ktl * 32 + k8 * 8 + j][stile * 16 + s]);
    const size_t base =
        ((((size_t)b * 512 + kt0 + ktl) * 4 + stile) * 512) + s * 32 + k8 * 8;
    *(uint4*)(swAp + base) = *(const uint4*)tmp;
  }
  __syncthreads();   // sws dead from here; its LDS is reused as Wqkv panels

  // ---- qkv = h @ Wqkv + bqkv; weight panels TDM-staged & shared by waves ----
#if HAVE_TDM
  if (wave == 0) tdm_load((unsigned)(size_t)&wpanel[0][0], Wqkvp, 1024); // 8 KB
  int cur = 0;
#endif
  for (int nt = 0; nt < 48; ++nt) {
#if HAVE_TDM
    if (wave == 0) __builtin_amdgcn_s_wait_tensorcnt(0);
    __syncthreads();
    if (wave == 0 && nt + 1 < 48)
      tdm_load((unsigned)(size_t)&wpanel[cur ^ 1][0],
               Wqkvp + (size_t)(nt + 1) * 4096, 1024);
    const unsigned short* wbase = &wpanel[cur][0];
#else
    const unsigned short* wbase = Wqkvp + (((size_t)nt * 8) << 9);
#endif
    v8f qacc = vzero8();
#pragma unroll
    for (int ks = 0; ks < 8; ++ks) {
      v16bf a = load_afrag(&hA[wrow][ks * 32], 256);
      qacc = wmma_bf16(a, load_bfrag(wbase + ((size_t)ks << 9)), qacc);
    }
#pragma unroll
    for (int r = 0; r < 8; ++r)
      colT[wrow + r + mrow][lane & 15] = qacc[r];
    __syncthreads();
    {
      const int task = threadIdx.x;
      const int k8 = task & 3, col = (task >> 2) & 15, ktl = task >> 6;
      const float bias = bqkv[nt * 16 + col];
      alignas(16) unsigned short tmp[8];
#pragma unroll
      for (int j = 0; j < 8; ++j)
        tmp[j] = f2bf(colT[ktl * 32 + k8 * 8 + j][col] + bias);
      const int tensor = nt >> 4, hh = (nt >> 1) & 7, dt = nt & 1;
      const size_t base =
          (((((size_t)tensor * Bb + b) * Hh + hh) * 512 + kt0 + ktl) * 2 + dt) * 512 +
          col * 32 + k8 * 8;
      *(uint4*)(qkvP + base) = *(const uint4*)tmp;
    }
#if HAVE_TDM
    cur ^= 1;
#else
    __syncthreads();
#endif
  }
}

// ---------------- k2: qs/ks/vs = sw^T @ {q,k,v} / denom, per (b,h) ----------
__global__ __launch_bounds__(256) void k2_pool(
    const unsigned short* __restrict__ swAp, const unsigned short* __restrict__ qkvP,
    float* __restrict__ pooled) {
  __shared__ float denomLDS[64];
  const int b = blockIdx.x >> 3, h = blockIdx.x & 7;
  const int wave = threadIdx.x >> 5, lane = threadIdx.x & 31;
  const int stile = wave >> 1, dt = wave & 1;
  const int mrow = (lane >> 4) << 3;
  const v16bf ones = ones_frag();
  v8f acc[3];
#pragma unroll
  for (int t = 0; t < 3; ++t) acc[t] = vzero8();
  v8f accD = vzero8();

  const unsigned short* swBase = swAp + (size_t)b * 512 * 2048;
  const unsigned short* qBase[3];
#pragma unroll
  for (int t = 0; t < 3; ++t)
    qBase[t] = qkvP + ((((size_t)t * Bb + b) * Hh + h) * 512) * 1024;

#if HAVE_TDM
  __shared__ unsigned short panel[2][5120];   // [A:2048][B: 3*1024] x2 = 20 KB
  if (wave == 0) {
    const unsigned lbase = (unsigned)(size_t)&panel[0][0];
    tdm_load(lbase, swBase, 512);
#pragma unroll
    for (int t = 0; t < 3; ++t)
      tdm_load(lbase + 4096 + t * 2048, qBase[t], 256);
  }
  int cur = 0;
  for (int kt = 0; kt < 512; ++kt) {
    if (wave == 0) __builtin_amdgcn_s_wait_tensorcnt(0);
    __syncthreads();
    if (wave == 0 && kt + 1 < 512) {
      const unsigned lbase = (unsigned)(size_t)&panel[cur ^ 1][0];
      tdm_load(lbase, swBase + (size_t)(kt + 1) * 2048, 512);
#pragma unroll
      for (int t = 0; t < 3; ++t)
        tdm_load(lbase + 4096 + t * 2048, qBase[t] + (size_t)(kt + 1) * 1024, 256);
    }
    v16bf a = load_afrag(&panel[cur][stile * 512], 32);
    accD = wmma_bf16(a, ones, accD);            // denom column sums via WMMA
#pragma unroll
    for (int t = 0; t < 3; ++t)
      acc[t] = wmma_bf16(a, load_bfrag(&panel[cur][2048 + t * 1024 + dt * 512]),
                         acc[t]);
    cur ^= 1;
  }
#else
  for (int kt = 0; kt < 512; ++kt) {
    v16bf a = load_afrag(swBase + (size_t)kt * 2048 + stile * 512, 32);
    accD = wmma_bf16(a, ones, accD);
#pragma unroll
    for (int t = 0; t < 3; ++t)
      acc[t] = wmma_bf16(a, load_bfrag(qBase[t] + (size_t)kt * 1024 + dt * 512),
                         acc[t]);
  }
#endif

  if (dt == 0 && (lane & 15) == 0) {
#pragma unroll
    for (int r = 0; r < 8; ++r)
      denomLDS[stile * 16 + r + mrow] = accD[r] + 1e-9f;
  }
  __syncthreads();
#pragma unroll
  for (int r = 0; r < 8; ++r) {
    const int s = stile * 16 + r + mrow;
    const float inv = 1.f / denomLDS[s];
#pragma unroll
    for (int t = 0; t < 3; ++t)
      pooled[((((size_t)t * Bb + b) * Hh + h) * 64 + s) * 32 + dt * 16 + (lane & 15)] =
          acc[t][r] * inv;
  }
}

// ---------------- k3: tiny 64-token attention per (b,h) ---------------------
__global__ __launch_bounds__(64) void k3_attn(const float* __restrict__ pooled,
                                              unsigned short* __restrict__ Tmatp) {
  const int b = blockIdx.x >> 3, h = blockIdx.x & 7;
  __shared__ float qsL[64][32], ksL[64][32], vsL[64][32], tokL[64][32];
  __shared__ float attnL[64][65];
  const int t = threadIdx.x;
  const size_t base0 = (((size_t)0 * Bb + b) * Hh + h) * 64 * 32;
  const size_t base1 = (((size_t)1 * Bb + b) * Hh + h) * 64 * 32;
  const size_t base2 = (((size_t)2 * Bb + b) * Hh + h) * 64 * 32;
  for (int j = 0; j < 32; ++j) {
    qsL[t][j] = pooled[base0 + t * 32 + j];
    ksL[t][j] = pooled[base1 + t * 32 + j];
    vsL[t][j] = pooled[base2 + t * 32 + j];
  }
  __syncthreads();
  const float scale = 0.17677669529663687f;   // 1/sqrt(32)
  float mx = -1e30f;
  for (int u = 0; u < 64; ++u) {
    float d = 0.f;
#pragma unroll
    for (int j = 0; j < 32; ++j) d += qsL[t][j] * ksL[u][j];
    d *= scale;
    attnL[t][u] = d;
    mx = fmaxf(mx, d);
  }
  float sum = 0.f;
  for (int u = 0; u < 64; ++u) { float e = __expf(attnL[t][u] - mx); attnL[t][u] = e; sum += e; }
  const float inv = 1.f / sum;
  float tok[32];
#pragma unroll
  for (int j = 0; j < 32; ++j) tok[j] = 0.f;
  for (int u = 0; u < 64; ++u) {
    const float w = attnL[t][u] * inv;
#pragma unroll
    for (int j = 0; j < 32; ++j) tok[j] += w * vsL[u][j];
  }
#pragma unroll
  for (int j = 0; j < 32; ++j) tokL[t][j] = tok[j];
  __syncthreads();
  for (int task = t; task < 256; task += 64) {
    const int k8 = task & 3, col = (task >> 2) & 15, kt = (task >> 6) & 1, ntl = task >> 7;
    const int ntile = 2 * h + ntl;
    alignas(16) unsigned short tmp[8];
#pragma unroll
    for (int j = 0; j < 8; ++j)
      tmp[j] = f2bf(tokL[kt * 32 + k8 * 8 + j][ntl * 16 + col]);
    const size_t base = (((size_t)b * 16 + ntile) * 2 + kt) * 512 + col * 32 + k8 * 8;
    *(uint4*)(Tmatp + base) = *(const uint4*)tmp;
  }
}

// ---------------- k4: out = (sw @ Tmat) @ Wout + bout -----------------------
__global__ __launch_bounds__(128) void k4_out(
    const unsigned short* __restrict__ swN, const unsigned short* __restrict__ Tmatp,
    const unsigned short* __restrict__ Woutp, const float* __restrict__ bout,
    float* __restrict__ out) {
  __shared__ unsigned short iA[64][256];     // 32 KB
#if HAVE_TDM
  __shared__ unsigned short wpan[2][4096];   // 16 KB Wout panel double buffer
#endif
  const int wave = threadIdx.x >> 5, lane = threadIdx.x & 31;
  const int rows0 = blockIdx.x * 64;
  const int b = rows0 / Nn;
  const int wrow = wave * 16;
  const int mrow = (lane >> 4) << 3;
  {
    v8f acc[16];
#pragma unroll
    for (int nt = 0; nt < 16; ++nt) acc[nt] = vzero8();
    const unsigned short* swrow = swN + (size_t)(rows0 + wrow) * Ss;
#pragma unroll
    for (int ks = 0; ks < 2; ++ks) {
      v16bf a = load_afrag(swrow + ks * 32, Ss);
      v16bf bc = load_bfrag(Tmatp + (((size_t)b * 16) * 2 + ks) * 512);
#pragma unroll
      for (int nt = 0; nt < 16; ++nt) {
        v16bf bn = bc;
        if (nt < 15)
          bn = load_bfrag(Tmatp + (((size_t)b * 16 + nt + 1) * 2 + ks) * 512);
        acc[nt] = wmma_bf16(a, bc, acc[nt]);
        bc = bn;
      }
    }
#pragma unroll
    for (int nt = 0; nt < 16; ++nt) {
      const int col = nt * 16 + (lane & 15);
#pragma unroll
      for (int r = 0; r < 8; ++r)
        iA[wrow + r + mrow][col] = f2bf(acc[nt][r]);
    }
  }
  __syncthreads();

  // second GEMM: nt-outer with TDM-staged Wout panels shared by the 4 waves
#if HAVE_TDM
  if (wave == 0) tdm_load((unsigned)(size_t)&wpan[0][0], Woutp, 1024);  // 8 KB
  int cur = 0;
#endif
  for (int nt = 0; nt < 16; ++nt) {
#if HAVE_TDM
    if (wave == 0) __builtin_amdgcn_s_wait_tensorcnt(0);
    __syncthreads();
    if (wave == 0 && nt + 1 < 16)
      tdm_load((unsigned)(size_t)&wpan[cur ^ 1][0],
               Woutp + (size_t)(nt + 1) * 4096, 1024);
    const unsigned short* wbase = &wpan[cur][0];
#else
    const unsigned short* wbase = Woutp + (((size_t)nt * 8) << 9);
#endif
    v8f acc = vzero8();
#pragma unroll
    for (int ks = 0; ks < 8; ++ks) {
      v16bf a = load_afrag(&iA[wrow][ks * 32], 256);
      acc = wmma_bf16(a, load_bfrag(wbase + ((size_t)ks << 9)), acc);
    }
    const int col = nt * 16 + (lane & 15);
    const float bias = bout[col];
#pragma unroll
    for (int r = 0; r < 8; ++r)
      out[(size_t)(rows0 + wrow + r + mrow) * Cc + col] = acc[r] + bias;
#if HAVE_TDM
    cur ^= 1;
#endif
  }
}

// ---------------- workspace layout (bytes) ----------------------------------
constexpr size_t OFF_W0P    = 0;
constexpr size_t OFF_WSP    = 131072;
constexpr size_t OFF_WQKVP  = 163840;
constexpr size_t OFF_WOUTP  = 557056;
constexpr size_t OFF_SWN    = 688128;
constexpr size_t OFF_SWAP   = 9076736;
constexpr size_t OFF_QKVP   = 17465344;
constexpr size_t OFF_POOLED = 118128640;
constexpr size_t OFF_TMATP  = 118915072;

extern "C" void kernel_launch(void* const* d_in, const int* in_sizes, int n_in,
                              void* d_out, int out_size, void* d_ws, size_t ws_size,
                              hipStream_t stream) {
  (void)in_sizes; (void)n_in; (void)out_size; (void)ws_size;
  const float* x    = (const float*)d_in[0];
  const float* W0   = (const float*)d_in[1];
  const float* b0   = (const float*)d_in[2];
  const float* Ws   = (const float*)d_in[3];
  const float* bs   = (const float*)d_in[4];
  const float* Wqkv = (const float*)d_in[5];
  const float* bqkv = (const float*)d_in[6];
  const float* Wout = (const float*)d_in[7];
  const float* bout = (const float*)d_in[8];
  float* out = (float*)d_out;
  char* ws = (char*)d_ws;
  unsigned short* W0p    = (unsigned short*)(ws + OFF_W0P);
  unsigned short* Wsp    = (unsigned short*)(ws + OFF_WSP);
  unsigned short* Wqkvp  = (unsigned short*)(ws + OFF_WQKVP);
  unsigned short* Woutp  = (unsigned short*)(ws + OFF_WOUTP);
  unsigned short* swN    = (unsigned short*)(ws + OFF_SWN);
  unsigned short* swAp   = (unsigned short*)(ws + OFF_SWAP);
  unsigned short* qkvP   = (unsigned short*)(ws + OFF_QKVP);
  float*          pooled = (float*)(ws + OFF_POOLED);
  unsigned short* Tmatp  = (unsigned short*)(ws + OFF_TMATP);

  k0_pack<<<32, 256, 0, stream>>>(W0,   W0p,   256, 256);
  k0_pack<<<8,  256, 0, stream>>>(Ws,   Wsp,   256, 64);
  k0_pack<<<96, 256, 0, stream>>>(Wqkv, Wqkvp, 256, 768);
  k0_pack<<<32, 256, 0, stream>>>(Wout, Woutp, 256, 256);

  k1_fused<<<(Bb * Nn) / 64, 128, 0, stream>>>(x, b0, bs, bqkv, W0p, Wsp, Wqkvp,
                                               swN, swAp, qkvP);
  k2_pool<<<Bb * Hh, 256, 0, stream>>>(swAp, qkvP, pooled);
  k3_attn<<<Bb * Hh, 64, 0, stream>>>(pooled, Tmatp);
  k4_out<<<(Bb * Nn) / 64, 128, 0, stream>>>(swN, Tmatp, Woutp, bout, out);
}
